// NaiveSSEAttention_70617852280889
// MI455X (gfx1250) — matrix operand order repaired and checked by reading, compiled, not run on hardware
//
#include <hip/hip_runtime.h>
#include <math.h>

// ---------------------------------------------------------------------------
// NaiveSSEAttention for MI455X (gfx1250, wave32, WMMA)
//   D=512, P=64, K=4, C=32, B=8, T=1024  -> tokens M = 8192
// Strategy:
//   * Dense GEMMs via v_wmma_f32_16x16x32_f16, operands PRE-PACKED into the
//     CDNA5 WMMA VGPR layout -> all fragment traffic is B128.
//   * Double-buffered LDS staging of B tiles via gfx1250 async global->LDS
//     copies (ASYNCcnt), one barrier per K-step.
//   * Sequential scan: 1 workgroup per batch, 32MB state lives in L2.
// ---------------------------------------------------------------------------

typedef __attribute__((ext_vector_type(16))) _Float16 v16h;
typedef __attribute__((ext_vector_type(8)))  float    v8f;

#define D_MODEL 512
#define NPART   64
#define TOPK    4
#define CROWS   32
#define BB      8
#define TT      1024
#define NTOK    (BB * TT)   // 8192

// ---- gfx1250 async global->LDS copy (ASYNCcnt path), with fallback --------
#if defined(__has_builtin)
#  if __has_builtin(__builtin_amdgcn_global_load_async_to_lds_b128)
#    define ASYNC_LDS 1
#  endif
#endif
#ifndef ASYNC_LDS
#  define ASYNC_LDS 0
#endif

#if ASYNC_LDS
// builtin expects pointers to a 16-byte int vector (per compiler diagnostic)
typedef int v4i __attribute__((vector_size(16)));
typedef __attribute__((address_space(1))) v4i v4i_g;   // global
typedef __attribute__((address_space(3))) v4i v4i_l;   // LDS
__device__ inline void async_wait_all() {
#  if __has_builtin(__builtin_amdgcn_s_wait_asynccnt)
    __builtin_amdgcn_s_wait_asynccnt(0);
#  else
    asm volatile("s_wait_asynccnt 0x0" ::: "memory");
#  endif
}
#endif

// ---------------------------------------------------------------------------
__device__ inline unsigned int pack2h(float a, float b) {
    _Float16 ha = (_Float16)a, hb = (_Float16)b;
    unsigned short ua, ub;
    __builtin_memcpy(&ua, &ha, 2);
    __builtin_memcpy(&ub, &hb, 2);
    return (unsigned int)ua | ((unsigned int)ub << 16);
}

// ---- pack fp32 A[M,K] (row-major) into WMMA A-fragment order --------------
// dword index = ((mt*ktiles + kt)*32 + lane)*8 + j
__global__ void pack_a_kernel(const float* __restrict__ A,
                              unsigned int* __restrict__ P, int M, int K) {
    const int i = blockIdx.x * blockDim.x + threadIdx.x;
    const int total = (M >> 4) * (K >> 5) * 256;
    if (i >= total) return;
    const int j    = i & 7;
    const int lane = (i >> 3) & 31;
    const int tile = i >> 8;
    const int ktiles = K >> 5;
    const int kt = tile % ktiles;
    const int mt = tile / ktiles;
    const int h = lane >> 4, r = lane & 15;
    const int row = mt * 16 + r;
    const int kk  = kt * 32 + ((j & 3) << 1) + (h << 3) + ((j >> 2) << 4);
    P[i] = pack2h(A[(size_t)row * K + kk], A[(size_t)row * K + kk + 1]);
}

// ---- pack fp32 B[K,N] (row-major) into WMMA B-fragment order --------------
// dword index = ((kt*ntiles + nt)*32 + lane)*8 + j
__global__ void pack_b_kernel(const float* __restrict__ B,
                              unsigned int* __restrict__ P, int K, int N) {
    const int i = blockIdx.x * blockDim.x + threadIdx.x;
    const int total = (K >> 5) * (N >> 4) * 256;
    if (i >= total) return;
    const int j    = i & 7;
    const int lane = (i >> 3) & 31;
    const int tile = i >> 8;
    const int ntiles = N >> 4;
    const int nt = tile % ntiles;
    const int kt = tile / ntiles;
    const int col = nt * 16 + (lane & 15);
    const int k0  = kt * 32 + ((j & 3) << 1) + ((lane >> 4) << 3) + ((j >> 2) << 4);
    P[i] = pack2h(B[(size_t)k0 * N + col], B[(size_t)(k0 + 1) * N + col]);
}

// --------------------------- WMMA GEMM (packed) ----------------------------
// C[M,N] = A*B + bias. Block = 256 thr = 8 waves -> 128 x (16*NSUB) tile.
// Wave w owns M-subtile (blockM*8+w) and all NSUB N-subtiles, reusing one A
// fragment per K-step across NSUB WMMAs. B tiles double-buffered in LDS.
template <int NSUB>
__global__ void __launch_bounds__(256)
gemm_wmma_packed_kernel(const uint4* __restrict__ Apack,
                        const uint4* __restrict__ Bpack,
                        const float* __restrict__ bias,
                        float* __restrict__ C,
                        int M, int N, int K) {
    const int ktiles  = K >> 5;
    const int ntiles  = N >> 4;
    const int nBlocks = ntiles / NSUB;
    const int blockM  = blockIdx.x / nBlocks;
    const int nb      = blockIdx.x % nBlocks;
    const int waveId  = threadIdx.x >> 5;
    const int lane    = threadIdx.x & 31;
    const int mt      = blockM * 8 + waveId;

    __shared__ uint4 bsh[2][NSUB * 64];          // double-buffered B tiles
    const bool stager = threadIdx.x < NSUB * 64; // wave-uniform predicate

    auto stage = [&](int kt, int buf) {
        const uint4* bsrc = Bpack + ((size_t)kt * ntiles + nb * NSUB) * 64
                                  + threadIdx.x;
        if (stager) {
#if ASYNC_LDS
            __builtin_amdgcn_global_load_async_to_lds_b128(
                (v4i_g*)bsrc, (v4i_l*)&bsh[buf][threadIdx.x], 0, 0);
#else
            bsh[buf][threadIdx.x] = *bsrc;
#endif
        }
    };

    v8f acc[NSUB] = {};
    const uint4* arow = Apack + (size_t)mt * ktiles * 64 + lane * 2;

    stage(0, 0);

    for (int kt = 0; kt < ktiles; ++kt) {
        const int cur = kt & 1;
#if ASYNC_LDS
        async_wait_all();        // drain this wave's async copies (ASYNCcnt)
#endif
        __syncthreads();         // buffer `cur` fully staged, `cur^1` free

        if (kt + 1 < ktiles) {
            stage(kt + 1, cur ^ 1);
            __builtin_prefetch(Bpack + ((size_t)(kt + 1) * ntiles + nb * NSUB) * 64
                                   + threadIdx.x, 0, 0);   // global_prefetch_b8
        }

        // A fragment: 32 contiguous bytes per lane (2 x global b128)
        union frag { v16h v; uint4 u[2]; };
        frag a;
        a.u[0] = arow[(size_t)kt * 64];
        a.u[1] = arow[(size_t)kt * 64 + 1];

        // gather all B fragments first (overlap ds traffic with WMMA issue)
        frag bf[NSUB];
#pragma unroll
        for (int n = 0; n < NSUB; ++n) {
            bf[n].u[0] = bsh[cur][n * 64 + lane * 2];      // ds_load_b128
            bf[n].u[1] = bsh[cur][n * 64 + lane * 2 + 1];
        }
#pragma unroll
        for (int n = 0; n < NSUB; ++n)
            acc[n] = __builtin_amdgcn_wmma_f32_16x16x32_f16(
                         false, a.v, false, bf[n].v, (short)0, acc[n], false, false);
    }

    // epilogue: C/D layout -> VGPR i = row i (lanes<16) / i+8 (lanes>=16)
    const int h = lane >> 4, r = lane & 15;
#pragma unroll
    for (int n = 0; n < NSUB; ++n) {
        const int col = (nb * NSUB + n) * 16 + r;
        const float bv = bias[col];
#pragma unroll
        for (int i = 0; i < 8; ++i) {
            const int row = mt * 16 + h * 8 + i;
            C[(size_t)row * N + col] = acc[n][i] + bv;
        }
    }
}

// ---------------- per-token top-4 selection + key softmax ------------------
__global__ void select_kernel(const float* __restrict__ logits,  // [NTOK,64]
                              const float* __restrict__ kproj,   // [NTOK,32]
                              int* __restrict__ idx_out,         // [NTOK,4]
                              float* __restrict__ w_out) {       // [NTOK,32]
    const int t = blockIdx.x * blockDim.x + threadIdx.x;
    if (t >= NTOK) return;

    const float* lg = logits + (size_t)t * NPART;
    unsigned long long taken = 0ull;
#pragma unroll
    for (int k = 0; k < TOPK; ++k) {
        float best = -__builtin_inff();
        int bi = 0;
        for (int p = 0; p < NPART; ++p) {
            if ((taken >> p) & 1ull) continue;
            float val = lg[p];
            if (val > best) { best = val; bi = p; }   // ties -> lowest index
        }
        taken |= 1ull << bi;
        idx_out[t * TOPK + k] = bi;
    }

    const float* kp = kproj + (size_t)t * CROWS;
    float m = -__builtin_inff();
#pragma unroll
    for (int c = 0; c < CROWS; ++c) m = fmaxf(m, kp[c]);
    float e[CROWS];
    float s = 0.0f;
#pragma unroll
    for (int c = 0; c < CROWS; ++c) { e[c] = __expf(kp[c] - m); s += e[c]; }
    const float inv = 1.0f / (s * (float)TOPK);
#pragma unroll
    for (int c = 0; c < CROWS; ++c) w_out[(size_t)t * CROWS + c] = e[c] * inv;
}

// ------------------------------- zero scratch ------------------------------
__global__ void zero_f32_kernel(float* __restrict__ p, int n) {
    int i = blockIdx.x * blockDim.x + threadIdx.x;
    if (i < n) p[i] = 0.0f;
}

// ------------------------------- recurrent scan ----------------------------
// grid = B (8 blocks), block = 512 threads (16 waves). S: [B,P,C,D] f32.
__global__ void __launch_bounds__(512)
scan_kernel(const int* __restrict__ idx,    // [B,T,4]
            const float* __restrict__ w,    // [B,T,32]
            const float* __restrict__ v,    // [B,T,512]
            const float* __restrict__ q,    // [B,T,512]
            float* __restrict__ S,          // [B,64,32,512]
            float* __restrict__ y) {        // [B,T,512]
    const int b   = blockIdx.x;
    const int tid = threadIdx.x;

    __shared__ float q_s[D_MODEL];
    __shared__ float v_s[D_MODEL];
    __shared__ float w_s[CROWS];
    __shared__ float sc[TOPK * CROWS];
    __shared__ int   idx_s[TOPK];

    float* Sb = S + (size_t)b * NPART * CROWS * D_MODEL;

    for (int t = 0; t < TT; ++t) {
        const size_t tok = (size_t)b * TT + t;
        q_s[tid] = q[tok * D_MODEL + tid];
        v_s[tid] = v[tok * D_MODEL + tid];
        if (tid < CROWS) w_s[tid]   = w[tok * CROWS + tid];
        if (tid < TOPK)  idx_s[tid] = idx[tok * TOPK + tid];
        __syncthreads();

        // phase 1: rank-1 state update + score partials (4 threads / (k,c))
        {
            const int kc  = tid >> 2;
            const int sub = tid & 3;
            const int k = kc >> 5, c = kc & (CROWS - 1);
            const int p = idx_s[k];
            const float wv = w_s[c];
            float part = 0.0f;
            float* row = Sb + ((size_t)p * CROWS + c) * D_MODEL;
            const int d0 = sub * 128;
#pragma unroll 8
            for (int d = d0; d < d0 + 128; ++d) {
                const float g = row[d] + wv * v_s[d];
                row[d] = g;
                part += g * q_s[d];
            }
            part += __shfl_down(part, 1, 4);
            part += __shfl_down(part, 2, 4);
            if (sub == 0) sc[kc] = part;
        }
        __syncthreads();

        // phase 2: softmax over 128 scores (wave 0)
        if (tid < 32) {
            float s0 = sc[tid], s1 = sc[tid + 32], s2 = sc[tid + 64], s3 = sc[tid + 96];
            float m = fmaxf(fmaxf(s0, s1), fmaxf(s2, s3));
#pragma unroll
            for (int off = 16; off; off >>= 1) m = fmaxf(m, __shfl_xor(m, off, 32));
            float e0 = __expf(s0 - m), e1 = __expf(s1 - m);
            float e2 = __expf(s2 - m), e3 = __expf(s3 - m);
            float sum = e0 + e1 + e2 + e3;
#pragma unroll
            for (int off = 16; off; off >>= 1) sum += __shfl_xor(sum, off, 32);
            const float inv = 1.0f / sum;
            sc[tid] = e0 * inv; sc[tid + 32] = e1 * inv;
            sc[tid + 64] = e2 * inv; sc[tid + 96] = e3 * inv;
        }
        __syncthreads();

        // phase 3: out[d] = sum_{k,c} attn * S[b,idx_k,c,d]  (L2 re-read)
        {
            float acc = 0.0f;
            const int d = tid;
#pragma unroll
            for (int k = 0; k < TOPK; ++k) {
                const int p = idx_s[k];
                const float* row = Sb + (size_t)p * CROWS * D_MODEL + d;
#pragma unroll 8
                for (int c = 0; c < CROWS; ++c)
                    acc += sc[k * CROWS + c] * row[(size_t)c * D_MODEL];
            }
            y[tok * D_MODEL + d] = acc;
        }
        __syncthreads();
    }
}

// ---------------------------------------------------------------------------
static inline int ceil_div(int a, int b) { return (a + b - 1) / b; }

extern "C" void kernel_launch(void* const* d_in, const int* in_sizes, int n_in,
                              void* d_out, int out_size, void* d_ws, size_t ws_size,
                              hipStream_t stream) {
    const float* x    = (const float*)d_in[0];
    const float* Wq   = (const float*)d_in[1];
    const float* bq   = (const float*)d_in[2];
    const float* Wk   = (const float*)d_in[3];
    const float* bk   = (const float*)d_in[4];
    const float* Wv   = (const float*)d_in[5];
    const float* bv   = (const float*)d_in[6];
    const float* Wo   = (const float*)d_in[7];
    const float* bo   = (const float*)d_in[8];
    const float* Wsel = (const float*)d_in[9];
    const float* bsel = (const float*)d_in[10];
    float* out = (float*)d_out;
    (void)ws_size; (void)in_sizes; (void)n_in; (void)out_size;

    // ---- workspace carving (~100 MB)
    char* base = (char*)d_ws;
    size_t off = 0;
    auto carve = [&](size_t bytes) { char* p = base + off;
                                     off += (bytes + 255) & ~(size_t)255; return p; };
    unsigned int* xpack  = (unsigned int*)carve((size_t)NTOK * D_MODEL / 2 * 4);
    unsigned int* Wq_p   = (unsigned int*)carve((size_t)D_MODEL * D_MODEL / 2 * 4);
    unsigned int* Wv_p   = (unsigned int*)carve((size_t)D_MODEL * D_MODEL / 2 * 4);
    unsigned int* Wo_p   = (unsigned int*)carve((size_t)D_MODEL * D_MODEL / 2 * 4);
    unsigned int* Wk_p   = (unsigned int*)carve((size_t)D_MODEL * CROWS / 2 * 4);
    unsigned int* Ws_p   = (unsigned int*)carve((size_t)D_MODEL * NPART / 2 * 4);
    unsigned int* ypack  = (unsigned int*)carve((size_t)NTOK * D_MODEL / 2 * 4);
    float* qbuf   = (float*)carve((size_t)NTOK * D_MODEL * 4);
    float* vbuf   = (float*)carve((size_t)NTOK * D_MODEL * 4);
    float* kproj  = (float*)carve((size_t)NTOK * CROWS * 4);
    float* logits = (float*)carve((size_t)NTOK * NPART * 4);
    float* wbuf   = (float*)carve((size_t)NTOK * CROWS * 4);
    int*   idxbuf = (int*)  carve((size_t)NTOK * TOPK * 4);
    float* Sbuf   = (float*)carve((size_t)BB * NPART * CROWS * D_MODEL * 4);
    float* ybuf   = (float*)carve((size_t)NTOK * D_MODEL * 4);

    // ---- pack activations + weights into WMMA fragment order (f32->f16)
    {
        const int blk = 256;
        int n;
        n = NTOK * D_MODEL / 2;
        pack_a_kernel<<<ceil_div(n, blk), blk, 0, stream>>>(x, xpack, NTOK, D_MODEL);
        n = D_MODEL * D_MODEL / 2;
        pack_b_kernel<<<ceil_div(n, blk), blk, 0, stream>>>(Wq, Wq_p, D_MODEL, D_MODEL);
        pack_b_kernel<<<ceil_div(n, blk), blk, 0, stream>>>(Wv, Wv_p, D_MODEL, D_MODEL);
        pack_b_kernel<<<ceil_div(n, blk), blk, 0, stream>>>(Wo, Wo_p, D_MODEL, D_MODEL);
        n = D_MODEL * CROWS / 2;
        pack_b_kernel<<<ceil_div(n, blk), blk, 0, stream>>>(Wk, Wk_p, D_MODEL, CROWS);
        n = D_MODEL * NPART / 2;
        pack_b_kernel<<<ceil_div(n, blk), blk, 0, stream>>>(Wsel, Ws_p, D_MODEL, NPART);
    }

    // ---- projection GEMMs
    auto gemm4 = [&](const unsigned int* Ap, const unsigned int* Bp,
                     const float* bias, float* C, int M, int N, int K) {
        const int blocks = (M / 128) * (N / 64);
        gemm_wmma_packed_kernel<4><<<blocks, 256, 0, stream>>>(
            (const uint4*)Ap, (const uint4*)Bp, bias, C, M, N, K);
    };
    auto gemm2 = [&](const unsigned int* Ap, const unsigned int* Bp,
                     const float* bias, float* C, int M, int N, int K) {
        const int blocks = (M / 128) * (N / 32);
        gemm_wmma_packed_kernel<2><<<blocks, 256, 0, stream>>>(
            (const uint4*)Ap, (const uint4*)Bp, bias, C, M, N, K);
    };
    gemm4(xpack, Wq_p, bq,   qbuf,   NTOK, D_MODEL, D_MODEL);
    gemm4(xpack, Wv_p, bv,   vbuf,   NTOK, D_MODEL, D_MODEL);
    gemm2(xpack, Wk_p, bk,   kproj,  NTOK, CROWS,   D_MODEL);
    gemm4(xpack, Ws_p, bsel, logits, NTOK, NPART,   D_MODEL);

    // ---- top-4 + key softmax
    select_kernel<<<ceil_div(NTOK, 256), 256, 0, stream>>>(logits, kproj, idxbuf, wbuf);

    // ---- zero state (deterministic per launch), then sequential scan
    {
        const int n = BB * NPART * CROWS * D_MODEL;
        zero_f32_kernel<<<ceil_div(n, 256), 256, 0, stream>>>(Sbuf, n);
    }
    scan_kernel<<<BB, 512, 0, stream>>>(idxbuf, wbuf, vbuf, qbuf, Sbuf, ybuf);

    // ---- output projection: out = y @ Wo + bo
    {
        const int n = NTOK * D_MODEL / 2;
        pack_a_kernel<<<ceil_div(n, 256), 256, 0, stream>>>(ybuf, ypack, NTOK, D_MODEL);
    }
    gemm4(ypack, Wo_p, bo, out, NTOK, D_MODEL, D_MODEL);
}
